// VisionRetentionChunk_26688926777442
// MI455X (gfx1250) — matrix operand order repaired and checked by reading, not compiled, not run on hardware
//
#include <hip/hip_runtime.h>
#include <hip/hip_bf16.h>

// ---------------------------------------------------------------------------
// CDNA5 (gfx1250) axial attention, bf16 WMMA path + TDM-staged attention K.
// Workspace layout (bf16 elements), total ~390 MB:
//   xb   [50176*768]   x in bf16 (reused later for the attention output)
//   qb   [50176*768]   q projection (bf16)
//   kb   [50176*768]   k projection, pre-scaled by KEY_DIM^-0.5 (bf16)
//   vb   [50176*768]   v projection (bf16)
//   vwb  [50176*768]   width-attention result, [b][h][n][x][d] (bf16)
//   Wq/Wk/Wv/Wo bf16   4 * 768*768
// ---------------------------------------------------------------------------

#define EMBED   768
#define NHEADS  12
#define KDIM    64
#define BB      16
#define HH      56
#define WW      56
#define MTOT    (BB * HH * WW)          // 50176
#define SCALING 0.125f                  // 64^-0.5

typedef __attribute__((ext_vector_type(16))) __bf16 v16bf;
typedef __attribute__((ext_vector_type(8)))  __bf16 v8bf;
typedef __attribute__((ext_vector_type(4)))  __bf16 v4bf;
typedef __attribute__((ext_vector_type(8)))  float  v8f;
typedef unsigned int u32x4 __attribute__((ext_vector_type(4)));
typedef int          i32x4 __attribute__((ext_vector_type(4)));
typedef int          i32x8 __attribute__((ext_vector_type(8)));

#if defined(__has_builtin)
#if __has_builtin(__builtin_amdgcn_tensor_load_to_lds) && \
    __has_builtin(__builtin_amdgcn_s_wait_tensorcnt)
#define HAVE_TDM 1
#endif
#endif
#ifndef HAVE_TDM
#define HAVE_TDM 0
#endif

__device__ __forceinline__ v16bf cat16(v8bf lo, v8bf hi) {
    return __builtin_shufflevector(lo, hi, 0,1,2,3,4,5,6,7,8,9,10,11,12,13,14,15);
}

__device__ __forceinline__ v8f wmma_bf16(v16bf a, v16bf b, v8f c) {
    return __builtin_amdgcn_wmma_f32_16x16x32_bf16(
        /*neg_a=*/false, a, /*neg_b=*/false, b,
        /*c_mod=*/(short)0, c, /*reuse_a=*/false, /*reuse_b=*/false);
}

// ---------------------------------------------------------------------------
// f32 -> bf16, 4 elements per thread (all sizes are multiples of 4)
// ---------------------------------------------------------------------------
__global__ void f32_to_bf16_v4(const float* __restrict__ in,
                               __bf16* __restrict__ out, int n4) {
    int i = blockIdx.x * blockDim.x + threadIdx.x;
    if (i < n4) {
        float4 f = ((const float4*)in)[i];
        v4bf o = { (__bf16)f.x, (__bf16)f.y, (__bf16)f.z, (__bf16)f.w };
        ((v4bf*)out)[i] = o;
    }
}

// ---------------------------------------------------------------------------
// C[M,768] = (A[M,768] @ W[768,768]^T + bias) * scale, all dims compile-time.
// Block: 128 threads = 4 waves; each wave owns a 32x32 C tile (2x2 WMMA tiles),
// block covers 64x64. Grid: (768/64, M/64).
// Fragment layouts per CDNA5 ISA:
//   A 16x32 bf16: lane(l): row = l&15; K = {kb..kb+7} u {kb+16..kb+23}, kb=8*(l>>4)
//   B 32x16 bf16: lane(l): col = l&15; K = {kb..kb+7} u {kb+16..kb+23} (same as A)
//   C  f32 16x16: vgpr r, lane l: row = r + 8*(l>>4), col = l&15
// ---------------------------------------------------------------------------
template <bool OUT_BF16>
__global__ __launch_bounds__(128)
void gemm768(const __bf16* __restrict__ A, const __bf16* __restrict__ Wt,
             const float* __restrict__ bias, float scale,
             __bf16* __restrict__ outb, float* __restrict__ outf) {
    constexpr int N = EMBED, K = EMBED;

    const int tid  = threadIdx.x;
    const int wave = tid >> 5;
    const int lane = tid & 31;
    const int l16  = lane & 15;
    const int lh   = lane >> 4;

    const int m0 = blockIdx.y * 64 + (wave >> 1) * 32;
    const int n0 = blockIdx.x * 64 + (wave & 1) * 32;

    v8f acc[2][2] = {};
    for (int kt = 0; kt < K; kt += 32) {
        v16bf a[2], b[2];
#pragma unroll
        for (int mi = 0; mi < 2; ++mi) {
            const __bf16* p = A + (size_t)(m0 + mi * 16 + l16) * K + kt + lh * 8;
            __builtin_prefetch(p + 64, 0, 1);       // global_prefetch_b8
            a[mi] = cat16(*(const v8bf*)p, *(const v8bf*)(p + 16));
        }
#pragma unroll
        for (int ni = 0; ni < 2; ++ni) {
            const __bf16* p = Wt + (size_t)(n0 + ni * 16 + l16) * K + kt + lh * 8;
            b[ni] = cat16(*(const v8bf*)p, *(const v8bf*)(p + 16));
        }
#pragma unroll
        for (int mi = 0; mi < 2; ++mi)
#pragma unroll
            for (int ni = 0; ni < 2; ++ni)
                acc[mi][ni] = wmma_bf16(a[mi], b[ni], acc[mi][ni]);
    }

    // Epilogue: constant strides -> base pointer + literal offsets.
#pragma unroll
    for (int mi = 0; mi < 2; ++mi)
#pragma unroll
        for (int ni = 0; ni < 2; ++ni) {
            const int   col  = n0 + ni * 16 + l16;
            const int   row0 = m0 + mi * 16 + lh * 8;
            const float bcol = bias[col];
            if (OUT_BF16) {
                __bf16* p = outb + (size_t)row0 * N + col;
#pragma unroll
                for (int r = 0; r < 8; ++r)
                    p[r * N] = (__bf16)((acc[mi][ni][r] + bcol) * scale);
            } else {
                float* p = outf + (size_t)row0 * N + col;
#pragma unroll
                for (int r = 0; r < 8; ++r)
                    p[r * N] = (acc[mi][ni][r] + bcol) * scale;
            }
        }
}

// ---------------------------------------------------------------------------
// Axial attention for one (batch, line, head): softmax(Q K^T) V over a 56-long
// axis, padded to 64. Scale is folded into the k projection.
// Block: 128 threads = 4 waves; wave t owns query rows [16t, 16t+16).
// STAGE 1: width attention (line = height row),  out = vw [b][h][n][x][d]
// STAGE 2: height attention (line = width col),  V = vw,  out = attn (b,x,w,e)
// K is staged into LDS by the Tensor Data Mover (one 2D descriptor per block,
// 56x64 bf16 tile, hardware zero-pads rows 56..63 via the tensor OOB rule).
// ---------------------------------------------------------------------------
template <int STAGE>
__global__ __launch_bounds__(128)
void axis_attention(const __bf16* __restrict__ q, const __bf16* __restrict__ k,
                    const __bf16* __restrict__ v, __bf16* __restrict__ out) {
    __shared__ __bf16 Ksm[64 * 64];        // [y][d]    row-major (TDM target)
    __shared__ __bf16 Vt [64 * 64];        // [d][y]    transposed
    __shared__ __bf16 Psm[4 * 16 * 64];    // per-wave probabilities [row][y]

    const int tid  = threadIdx.x;
    const int wave = tid >> 5;
    const int lane = tid & 31;
    const int l16  = lane & 15;
    const int lh   = lane >> 4;

    int p    = blockIdx.x;
    int head = p % NHEADS;
    int line = (p / NHEADS) % HH;
    int bb   = p / (NHEADS * HH);

    constexpr int qkRow  = (STAGE == 1) ? EMBED : WW * EMBED;
    constexpr int vRow   = (STAGE == 1) ? EMBED : NHEADS * WW * KDIM;
    constexpr int outRow = (STAGE == 1) ? KDIM  : WW * EMBED;

    size_t qkBase, vBase, outBase;
    if (STAGE == 1) {           // width axis: x,y run over W at fixed height=line
        qkBase  = ((size_t)(bb * HH + line) * WW) * EMBED + head * KDIM;
        vBase   = qkBase;
        outBase = (((size_t)(bb * HH + line) * NHEADS + head) * WW) * KDIM;
    } else {                    // height axis: x,y run over H at fixed width=line
        qkBase  = ((size_t)(bb * HH) * WW + line) * EMBED + head * KDIM;
        vBase   = (((size_t)(bb * HH) * NHEADS + head) * WW + line) * KDIM;
        outBase = qkBase;
    }

#if HAVE_TDM
    // --- K tile via Tensor Data Mover: one 2D descriptor, issued by wave 0 ---
    // D# group layouts per CDNA5 ISA.
    if (wave == 0) {
        const unsigned long long ga = (unsigned long long)(size_t)(k + qkBase);
        const unsigned int lds = (unsigned int)(size_t)(&Ksm[0]); // LDS byte offset
        u32x4 g0 = { 1u,                                    // count=1, user D#
                     lds,                                   // lds_addr
                     (unsigned int)ga,                      // global_addr[31:0]
                     (unsigned int)((ga >> 32) & 0x1FFFFFFu) | 0x80000000u }; // [56:32] | type=2
        const unsigned long long tdim0 = 64, tdim1 = HH;    // tensor: 56 rows x 64 elems
        const unsigned long long tile0 = 64, tile1 = 64;    // tile: 64 rows -> OOB rows = 0
        const unsigned long long str0  = qkRow;             // row stride (elements)
        const unsigned long long q0 = (1ull << 16)                    // data_size = 2 bytes
                                    | ((tdim0 & 0xFFFFull) << 48);    // tensor_dim0[15:0]
        const unsigned long long q1 = (tdim0 >> 16)                   // tensor_dim0[31:16]
                                    | (tdim1 << 16)                   // tensor_dim1
                                    | (tile0 << 48);                  // tile_dim0
        const unsigned long long q2 = tile1                           // tile_dim1
                                    | ((str0 & 0xFFFFFFFFull) << 32); // dim0_stride[31:0]
        const unsigned long long q3 = (str0 >> 32) & 0xFFFFull;       // dim0_stride[47:32]
        i32x8 g1 = { (int)(unsigned)q0, (int)(unsigned)(q0 >> 32),
                     (int)(unsigned)q1, (int)(unsigned)(q1 >> 32),
                     (int)(unsigned)q2, (int)(unsigned)(q2 >> 32),
                     (int)(unsigned)q3, (int)(unsigned)(q3 >> 32) };
        i32x4 gz4 = { 0, 0, 0, 0 };                         // groups 2/3: dims unused
        i32x8 gz8 = { 0, 0, 0, 0, 0, 0, 0, 0 };             // 6-arg form extra group
        __builtin_amdgcn_tensor_load_to_lds(g0, g1, gz4, gz4, gz8, 0);
        __builtin_amdgcn_s_wait_tensorcnt(0);
    }
    // V staged (transposed + zero-padded) by all threads.
    for (int idx = tid; idx < 64 * 64; idx += 128) {
        int y = idx >> 6, c = idx & 63;
        Vt[c * 64 + y] = (y < HH) ? v[vBase + (size_t)y * vRow + c] : (__bf16)0.0f;
    }
#else
    // Fallback: cooperative stage of K (row-major) and V (transposed).
    for (int idx = tid; idx < 64 * 64; idx += 128) {
        int y = idx >> 6, c = idx & 63;
        Ksm[idx]       = (y < HH) ? k[qkBase + (size_t)y * qkRow + c] : (__bf16)0.0f;
        Vt[c * 64 + y] = (y < HH) ? v[vBase  + (size_t)y * vRow  + c] : (__bf16)0.0f;
    }
#endif
    __syncthreads();

    // ---- S = Q K^T : 4 column tiles, K-dim = KDIM = 64 (2 steps) ----
    v8f accS[4] = {};
    const int xA = wave * 16 + l16;                 // A-fragment query row
    for (int kt = 0; kt < KDIM; kt += 32) {
        v8bf qlo = {}, qhi = {};
        if (xA < HH) {
            const __bf16* qp = q + qkBase + (size_t)xA * qkRow + kt + lh * 8;
            qlo = *(const v8bf*)qp;
            qhi = *(const v8bf*)(qp + 16);
        }
        v16bf aq = cat16(qlo, qhi);
#pragma unroll
        for (int ty = 0; ty < 4; ++ty) {
            const __bf16* kp = Ksm + (ty * 16 + l16) * 64 + kt + lh * 8;
            accS[ty] = wmma_bf16(aq, cat16(*(const v8bf*)kp, *(const v8bf*)(kp + 16)),
                                 accS[ty]);
        }
    }

    // Mask padded key columns (tile 3, cols 56..63) to -inf.
    if (l16 >= 8) {
#pragma unroll
        for (int r = 0; r < 8; ++r) accS[3][r] = -1e30f;
    }

    // ---- softmax over y, per C row; rows live in 16-lane halves ----
#pragma unroll
    for (int r = 0; r < 8; ++r) {
        float s0 = accS[0][r], s1 = accS[1][r], s2 = accS[2][r], s3 = accS[3][r];
        float mx = fmaxf(fmaxf(s0, s1), fmaxf(s2, s3));
#pragma unroll
        for (int off = 8; off; off >>= 1) mx = fmaxf(mx, __shfl_xor(mx, off, 16));
        float e0 = __expf(s0 - mx), e1 = __expf(s1 - mx);
        float e2 = __expf(s2 - mx), e3 = __expf(s3 - mx);
        float sum = e0 + e1 + e2 + e3;
#pragma unroll
        for (int off = 8; off; off >>= 1) sum += __shfl_xor(sum, off, 16);
        float inv = 1.0f / sum;
        int row = r + lh * 8;
        __bf16* pr = Psm + wave * 1024 + row * 64 + l16;
        pr[0]  = (__bf16)(e0 * inv);
        pr[16] = (__bf16)(e1 * inv);
        pr[32] = (__bf16)(e2 * inv);
        pr[48] = (__bf16)(e3 * inv);
    }
    // Psm is wave-private; LDS ops are in-order within a wave -> no barrier.

    // ---- O = P V : 4 output tiles over d, K-dim = 64 padded y (2 steps) ----
    v8f accO[4] = {};
    for (int kt = 0; kt < 64; kt += 32) {
        const __bf16* pp = Psm + wave * 1024 + l16 * 64 + kt + lh * 8;
        v16bf ap = cat16(*(const v8bf*)pp, *(const v8bf*)(pp + 16));
#pragma unroll
        for (int td = 0; td < 4; ++td) {
            const __bf16* vp = Vt + (td * 16 + l16) * 64 + kt + lh * 8;
            accO[td] = wmma_bf16(ap, cat16(*(const v8bf*)vp, *(const v8bf*)(vp + 16)),
                                 accO[td]);
        }
    }

    // Store: constant outRow -> literal offsets off a per-lane base pointer.
    {
        const int x0 = wave * 16 + lh * 8;          // first of this half's 8 rows
        __bf16* po = out + outBase + (size_t)x0 * outRow + l16;
#pragma unroll
        for (int td = 0; td < 4; ++td) {
#pragma unroll
            for (int r = 0; r < 8; ++r) {
                if (x0 + r < HH)
                    po[r * outRow + td * 16] = (__bf16)accO[td][r];
            }
        }
    }
}

// ---------------------------------------------------------------------------
extern "C" void kernel_launch(void* const* d_in, const int* in_sizes, int n_in,
                              void* d_out, int out_size, void* d_ws, size_t ws_size,
                              hipStream_t stream) {
    const float* x  = (const float*)d_in[0];
    const float* Wq = (const float*)d_in[1];
    const float* bq = (const float*)d_in[2];
    const float* Wk = (const float*)d_in[3];
    const float* bk = (const float*)d_in[4];
    const float* Wv = (const float*)d_in[5];
    const float* bv = (const float*)d_in[6];
    const float* Wo = (const float*)d_in[7];
    const float* bo = (const float*)d_in[8];
    float* outp = (float*)d_out;

    const size_t SZ_X = (size_t)MTOT * EMBED;       // 38,535,168 elements
    const size_t SZ_W = (size_t)EMBED * EMBED;      // 589,824 elements

    __bf16* xb  = (__bf16*)d_ws;                    // also reused as attn output
    __bf16* qb  = xb  + SZ_X;
    __bf16* kb  = qb  + SZ_X;
    __bf16* vb  = kb  + SZ_X;
    __bf16* vwb = vb  + SZ_X;
    __bf16* wqb = vwb + SZ_X;
    __bf16* wkb = wqb + SZ_W;
    __bf16* wvb = wkb + SZ_W;
    __bf16* wob = wvb + SZ_W;

    // 1) fp32 -> bf16 converts
    {
        int n4 = (int)(SZ_X / 4);
        f32_to_bf16_v4<<<(n4 + 255) / 256, 256, 0, stream>>>(x, xb, n4);
        int w4 = (int)(SZ_W / 4);
        f32_to_bf16_v4<<<(w4 + 255) / 256, 256, 0, stream>>>(Wq, wqb, w4);
        f32_to_bf16_v4<<<(w4 + 255) / 256, 256, 0, stream>>>(Wk, wkb, w4);
        f32_to_bf16_v4<<<(w4 + 255) / 256, 256, 0, stream>>>(Wv, wvb, w4);
        f32_to_bf16_v4<<<(w4 + 255) / 256, 256, 0, stream>>>(Wo, wob, w4);
    }

    // 2) q/k/v projections (scale folded into k)
    dim3 ggrid(EMBED / 64, MTOT / 64);              // (12, 784)
    gemm768<true><<<ggrid, 128, 0, stream>>>(xb, wqb, bq, 1.0f,    qb, nullptr);
    gemm768<true><<<ggrid, 128, 0, stream>>>(xb, wkb, bk, SCALING, kb, nullptr);
    gemm768<true><<<ggrid, 128, 0, stream>>>(xb, wvb, bv, 1.0f,    vb, nullptr);

    // 3) width-axis attention -> vw;  4) height-axis attention -> attn (reuse xb)
    const int nprob = BB * HH * NHEADS;             // 10752
    axis_attention<1><<<nprob, 128, 0, stream>>>(qb, kb, vb,  vwb);
    axis_attention<2><<<nprob, 128, 0, stream>>>(qb, kb, vwb, xb);

    // 5) output projection -> f32 d_out
    gemm768<false><<<ggrid, 128, 0, stream>>>(xb, wob, bo, 1.0f, nullptr, outp);
}
